// LinearDecoderTF_55929064128631
// MI455X (gfx1250) — matrix operand order repaired and checked by reading, not compiled, not run on hardware
//
#include <hip/hip_runtime.h>
#include <stdint.h>

// ---------------------------------------------------------------------------
// LinearDecoderTF on MI455X (gfx1250): all GEMMs via v_wmma_f32_16x16x32_f16
// B=4096, LAT=128, MLP=HID=512, IN=16, T=32, GATE=4*H=2048
// ---------------------------------------------------------------------------

typedef __attribute__((ext_vector_type(16))) _Float16 v16h;
typedef __attribute__((ext_vector_type(8)))  _Float16 v8h;
typedef __attribute__((ext_vector_type(8)))  float    v8f;

#define B_SZ 4096
#define LAT  128
#define HID  512
#define INSZ 16
#define TLEN 32
#define MLPW 512
#define GATE 2048
#define XPAD 32   // xt padded to K=32 so it is one WMMA K-tile

union H16U { v16h v; v8h h[2]; };
__device__ __forceinline__ v16h pack16(v8h lo, v8h hi) {
  H16U u; u.h[0] = lo; u.h[1] = hi; return u.v;
}
__device__ __forceinline__ float sigf(float x) { return 1.f / (1.f + __expf(-x)); }

// ------------------------------ prep kernels -------------------------------

__global__ __launch_bounds__(256) void k_cvt_f16(const float* __restrict__ s,
                                                 _Float16* __restrict__ d, int n) {
  int i = blockIdx.x * 256 + threadIdx.x;
  if (i < n) d[i] = (_Float16)s[i];
}

__global__ __launch_bounds__(256) void k_pad_wih(const float* __restrict__ w,
                                                 _Float16* __restrict__ d) {
  int i = blockIdx.x * 256 + threadIdx.x;          // over GATE*XPAD
  if (i < GATE * XPAD) {
    int n = i / XPAD, k = i % XPAD;
    d[i] = (k < INSZ) ? (_Float16)w[n * INSZ + k] : (_Float16)0.f;
  }
}

__global__ __launch_bounds__(256) void k_bias_sum(const float* __restrict__ a,
                                                  const float* __restrict__ b,
                                                  float* __restrict__ o, int n) {
  int i = blockIdx.x * 256 + threadIdx.x;
  if (i < n) o[i] = a[i] + b[i];
}

__global__ __launch_bounds__(256) void k_zero_h16(_Float16* __restrict__ d, int n) {
  int i = blockIdx.x * 256 + threadIdx.x;
  if (i < n) d[i] = (_Float16)0.f;
}

// ------------------------------ main WMMA GEMM -----------------------------
// out[M,N] = act(A[M,K] * W[N,K]^T + bias) ; optional A2[M,32]*W2[N,32]^T term.
// Wave tile 32x64 (2x4 wmma accumulators), block = 8 waves -> 64x256 tile.
// grid = (N/256, M/64), block = 256 threads. M,K multiples of 32; N of 256.

__global__ __launch_bounds__(256) void k_gemm(
    const _Float16* __restrict__ A, const _Float16* __restrict__ W,
    const float* __restrict__ bias,
    const _Float16* __restrict__ A2, const _Float16* __restrict__ W2,
    _Float16* __restrict__ out16, float* __restrict__ out32,
    int N, int K, int leaky) {
  const int lane  = threadIdx.x & 31;
  const int wave  = threadIdx.x >> 5;
  const int lhalf = lane & 15;   // N (B/C) column, M (A) row within tile
  const int lhi   = lane >> 4;   // K-half selector
  const int m0 = blockIdx.y * 64  + (wave >> 2) * 32;
  const int n0 = blockIdx.x * 256 + (wave & 3) * 64;

  v8f acc[2][4] = {};

  // per-lane base pointers; advance by 32 halves each K-step
  const _Float16* apb[2];
  const _Float16* bpb[4];
#pragma unroll
  for (int i = 0; i < 2; ++i)
    apb[i] = A + (size_t)(m0 + i * 16 + lhalf) * K + lhi * 16;
#pragma unroll
  for (int j = 0; j < 4; ++j)
    bpb[j] = W + (size_t)(n0 + j * 16 + lhalf) * K + lhi * 16;

  for (int k0 = 0; k0 < K; k0 += 32) {
    v16h af[2];
#pragma unroll
    for (int i = 0; i < 2; ++i) {
      const _Float16* ap = apb[i] + k0;
      af[i] = pack16(*(const v8h*)ap, *(const v8h*)(ap + 8));
      __builtin_prefetch(ap + 32, 0, 1);   // speculative: next K-tile
    }
    v16h bf[4];
#pragma unroll
    for (int j = 0; j < 4; ++j) {
      const _Float16* bp = bpb[j] + k0;
      bf[j] = *(const v16h*)bp;
      __builtin_prefetch(bp + 32, 0, 1);   // speculative: next K-tile
    }
#pragma unroll
    for (int i = 0; i < 2; ++i)
#pragma unroll
      for (int j = 0; j < 4; ++j)
        acc[i][j] = __builtin_amdgcn_wmma_f32_16x16x32_f16(
            false, af[i], false, bf[j], (short)0, acc[i][j], false, false);
  }

  if (A2) {  // single K=32 tile: xt_pad[M,32] * Wih_pad[N,32]^T
    v16h af[2];
#pragma unroll
    for (int i = 0; i < 2; ++i) {
      const _Float16* ap = A2 + (size_t)(m0 + i * 16 + lhalf) * XPAD + lhi * 16;
      af[i] = pack16(*(const v8h*)ap, *(const v8h*)(ap + 8));
    }
    v16h bf[4];
#pragma unroll
    for (int j = 0; j < 4; ++j) {
      const _Float16* bp = W2 + (size_t)(n0 + j * 16 + lhalf) * XPAD + lhi * 16;
      bf[j] = *(const v16h*)bp;
    }
#pragma unroll
    for (int i = 0; i < 2; ++i)
#pragma unroll
      for (int j = 0; j < 4; ++j)
        acc[i][j] = __builtin_amdgcn_wmma_f32_16x16x32_f16(
            false, af[i], false, bf[j], (short)0, acc[i][j], false, false);
  }

  // epilogue: bias + optional LeakyReLU, hoisted output-format branches
  float r[2][4][8];
#pragma unroll
  for (int j = 0; j < 4; ++j) {
    const float bn = bias[n0 + j * 16 + lhalf];
#pragma unroll
    for (int i = 0; i < 2; ++i)
#pragma unroll
      for (int q = 0; q < 8; ++q) {
        float v = acc[i][j][q] + bn;
        if (leaky) v = (v >= 0.f) ? v : 0.2f * v;
        r[i][j][q] = v;
      }
  }
  if (out32) {
#pragma unroll
    for (int i = 0; i < 2; ++i)
#pragma unroll
      for (int q = 0; q < 8; ++q) {
        const size_t row = (size_t)(m0 + i * 16 + lhi * 8 + q) * N;
#pragma unroll
        for (int j = 0; j < 4; ++j)
          out32[row + n0 + j * 16 + lhalf] = r[i][j][q];
      }
  }
  if (out16) {
#pragma unroll
    for (int i = 0; i < 2; ++i)
#pragma unroll
      for (int q = 0; q < 8; ++q) {
        const size_t row = (size_t)(m0 + i * 16 + lhi * 8 + q) * N;
#pragma unroll
        for (int j = 0; j < 4; ++j)
          out16[row + n0 + j * 16 + lhalf] = (_Float16)r[i][j][q];
      }
  }
}

// -------------------- N=16 projection + softmax/sigmoid head ---------------
// One wave computes a 16x16 tile (16 batch rows, all 16 outputs). K = 512.
// Row M's 16 outputs live in one 16-lane half of a VGPR -> shfl_xor softmax.

__global__ __launch_bounds__(256) void k_head(
    const _Float16* __restrict__ A,   // [B,512] f16 activations
    const _Float16* __restrict__ Ws,  // [16,512] f16
    const float* __restrict__ bias,   // [16]
    float* __restrict__ ys,           // d_out base [B,T,IN] or null
    int t,
    _Float16* __restrict__ xtpad) {   // [B,32] next-step LSTM input
  const int lane  = threadIdx.x & 31;
  const int wave  = threadIdx.x >> 5;
  const int lhalf = lane & 15;
  const int lhi   = lane >> 4;
  const int m0 = (blockIdx.x * 8 + wave) * 16;

  const _Float16* ap0 = A  + (size_t)(m0 + lhalf) * HID + lhi * 16;
  const _Float16* bp0 = Ws + (size_t)lhalf * HID + lhi * 16;

  v8f acc = {};
  for (int k0 = 0; k0 < HID; k0 += 32) {
    v16h af = pack16(*(const v8h*)(ap0 + k0), *(const v8h*)(ap0 + k0 + 8));
    v16h bf = *(const v16h*)(bp0 + k0);
    acc = __builtin_amdgcn_wmma_f32_16x16x32_f16(false, af, false, bf,
                                                 (short)0, acc, false, false);
  }

  const float bn = bias[lhalf];
#pragma unroll
  for (int q = 0; q < 8; ++q) {
    float v = acc[q] + bn;
    // softmax over n=0..14 (within this 16-lane half), sigmoid on n=15
    float mx = (lhalf < 15) ? v : -1e30f;
#pragma unroll
    for (int m = 1; m <= 8; m <<= 1) mx = fmaxf(mx, __shfl_xor(mx, m, 32));
    float e = (lhalf < 15) ? __expf(v - mx) : 0.f;
    float s = e;
#pragma unroll
    for (int m = 1; m <= 8; m <<= 1) s += __shfl_xor(s, m, 32);
    float o = (lhalf < 15) ? (e / s) : sigf(v);

    const int b = m0 + lhi * 8 + q;
    if (ys) ys[(size_t)b * (TLEN * INSZ) + t * INSZ + lhalf] = o;
    xtpad[b * XPAD + lhalf] = (_Float16)o;
  }
}

// ------------------------------ LSTM cell ----------------------------------

__global__ __launch_bounds__(256) void k_lstm_cell(
    const float* __restrict__ g,      // [B, 2048] gates (i,f,g,o)
    float* __restrict__ c,            // [B, 512] fp32 state (in/out)
    _Float16* __restrict__ h16) {     // [B, 512] f16 hidden for next GEMM
  int i = blockIdx.x * 256 + threadIdx.x;
  if (i >= B_SZ * HID) return;
  int b = i / HID, j = i % HID;
  const float* gr = g + (size_t)b * GATE;
  float si = sigf(gr[j]);
  float sf = sigf(gr[HID + j]);
  float tg = tanhf(gr[2 * HID + j]);
  float so = sigf(gr[3 * HID + j]);
  float cn = sf * c[i] + si * tg;
  c[i] = cn;
  h16[i] = (_Float16)(so * tanhf(cn));
}

// ------------------------------ host driver --------------------------------

extern "C" void kernel_launch(void* const* d_in, const int* in_sizes, int n_in,
                              void* d_out, int out_size, void* d_ws, size_t ws_size,
                              hipStream_t stream) {
  const float* x   = (const float*)d_in[0];
  const float* W1  = (const float*)d_in[1];  const float* b1  = (const float*)d_in[2];
  const float* W2  = (const float*)d_in[3];  const float* b2  = (const float*)d_in[4];
  const float* W3  = (const float*)d_in[5];  const float* b3  = (const float*)d_in[6];
  const float* Wh1 = (const float*)d_in[7];  const float* bh1 = (const float*)d_in[8];
  const float* Wh2 = (const float*)d_in[9];  const float* bh2 = (const float*)d_in[10];
  const float* Wc1 = (const float*)d_in[11]; const float* bc1 = (const float*)d_in[12];
  const float* Wc2 = (const float*)d_in[13]; const float* bc2 = (const float*)d_in[14];
  const float* Wx1 = (const float*)d_in[15]; const float* bx1 = (const float*)d_in[16];
  const float* Wx2 = (const float*)d_in[17]; const float* bx2 = (const float*)d_in[18];
  const float* Wih = (const float*)d_in[19]; const float* bih = (const float*)d_in[20];
  const float* Whh = (const float*)d_in[21]; const float* bhh = (const float*)d_in[22];
  const float* Wp  = (const float*)d_in[23]; const float* bp  = (const float*)d_in[24];
  float* out = (float*)d_out;

  // bump allocator over workspace (256B aligned carves)
  char* wp = (char*)d_ws;
  auto alloc = [&](size_t bytes) -> void* {
    uintptr_t a = ((uintptr_t)wp + 255) & ~(uintptr_t)255;
    wp = (char*)a + bytes;
    return (void*)a;
  };
  _Float16* hW1  = (_Float16*)alloc((size_t)MLPW * LAT  * 2);
  _Float16* hW2  = (_Float16*)alloc((size_t)MLPW * MLPW * 2);
  _Float16* hW3  = (_Float16*)alloc((size_t)MLPW * MLPW * 2);
  _Float16* hWh1 = (_Float16*)alloc((size_t)MLPW * MLPW * 2);
  _Float16* hWh2 = (_Float16*)alloc((size_t)HID  * MLPW * 2);
  _Float16* hWc1 = (_Float16*)alloc((size_t)MLPW * MLPW * 2);
  _Float16* hWc2 = (_Float16*)alloc((size_t)HID  * MLPW * 2);
  _Float16* hWx1 = (_Float16*)alloc((size_t)MLPW * MLPW * 2);
  _Float16* hWx2 = (_Float16*)alloc((size_t)INSZ * MLPW * 2);
  _Float16* hWhh = (_Float16*)alloc((size_t)GATE * HID  * 2);
  _Float16* hWihP= (_Float16*)alloc((size_t)GATE * XPAD * 2);
  _Float16* hWp  = (_Float16*)alloc((size_t)INSZ * HID  * 2);
  _Float16* hx   = (_Float16*)alloc((size_t)B_SZ * LAT  * 2);
  _Float16* hz   = (_Float16*)alloc((size_t)B_SZ * MLPW * 2);
  _Float16* ht   = (_Float16*)alloc((size_t)B_SZ * MLPW * 2);
  _Float16* hh   = (_Float16*)alloc((size_t)B_SZ * HID  * 2);
  _Float16* hxt  = (_Float16*)alloc((size_t)B_SZ * XPAD * 2);
  float*    fBg  = (float*)   alloc((size_t)GATE * 4);
  float*    fC   = (float*)   alloc((size_t)B_SZ * HID  * 4);
  float*    fG   = (float*)   alloc((size_t)B_SZ * GATE * 4);

  auto cvt = [&](const float* s, _Float16* d, int n) {
    k_cvt_f16<<<(n + 255) / 256, 256, 0, stream>>>(s, d, n);
  };
  // weight / input conversion to f16 (deterministic, re-done each call)
  cvt(x,   hx,   B_SZ * LAT);
  cvt(W1,  hW1,  MLPW * LAT);
  cvt(W2,  hW2,  MLPW * MLPW);
  cvt(W3,  hW3,  MLPW * MLPW);
  cvt(Wh1, hWh1, MLPW * MLPW);
  cvt(Wh2, hWh2, HID  * MLPW);
  cvt(Wc1, hWc1, MLPW * MLPW);
  cvt(Wc2, hWc2, HID  * MLPW);
  cvt(Wx1, hWx1, MLPW * MLPW);
  cvt(Wx2, hWx2, INSZ * MLPW);
  cvt(Whh, hWhh, GATE * HID);
  cvt(Wp,  hWp,  INSZ * HID);
  k_pad_wih<<<(GATE * XPAD + 255) / 256, 256, 0, stream>>>(Wih, hWihP);
  k_bias_sum<<<(GATE + 255) / 256, 256, 0, stream>>>(bih, bhh, fBg, GATE);
  k_zero_h16<<<(B_SZ * XPAD + 255) / 256, 256, 0, stream>>>(hxt, B_SZ * XPAD);

  auto gemm = [&](const _Float16* A, const _Float16* W, const float* bias,
                  const _Float16* A2, const _Float16* Wpad,
                  _Float16* o16, float* o32, int N, int K, int leaky) {
    dim3 grid(N / 256, B_SZ / 64);
    k_gemm<<<grid, 256, 0, stream>>>(A, W, bias, A2, Wpad, o16, o32, N, K, leaky);
  };

  // ---- front MLP ----
  gemm(hx, hW1, b1, nullptr, nullptr, hz, nullptr, MLPW, LAT,  1);  // z1
  gemm(hz, hW2, b2, nullptr, nullptr, ht, nullptr, MLPW, MLPW, 1);  // z2
  gemm(ht, hW3, b3, nullptr, nullptr, hz, nullptr, MLPW, MLPW, 1);  // z
  // h0
  gemm(hz, hWh1, bh1, nullptr, nullptr, ht, nullptr, MLPW, MLPW, 1);
  gemm(ht, hWh2, bh2, nullptr, nullptr, hh, nullptr, HID,  MLPW, 0);
  // c0
  gemm(hz, hWc1, bc1, nullptr, nullptr, ht, nullptr, MLPW, MLPW, 1);
  gemm(ht, hWc2, bc2, nullptr, nullptr, nullptr, fC, HID,  MLPW, 0);
  // x0 -> first xt (not in output)
  gemm(hz, hWx1, bx1, nullptr, nullptr, ht, nullptr, MLPW, MLPW, 1);
  k_head<<<B_SZ / 128, 256, 0, stream>>>(ht, hWx2, bx2, nullptr, 0, hxt);

  // ---- LSTM decode: T serial steps ----
  for (int t = 0; t < TLEN; ++t) {
    // g = h @ Whh^T + xt_pad @ WihPad^T + (bih+bhh)
    gemm(hh, hWhh, fBg, hxt, hWihP, nullptr, fG, GATE, HID, 0);
    k_lstm_cell<<<(B_SZ * HID) / 256, 256, 0, stream>>>(fG, fC, hh);
    k_head<<<B_SZ / 128, 256, 0, stream>>>(hh, hWp, bp, out, t, hxt);
  }
}
